// Fp8PTPCLinear_69879117906072
// MI455X (gfx1250) — 8-bit matrix operand order repaired and checked by reading, compiled, not run on hardware
//
#include <hip/hip_runtime.h>
#include <hip/hip_bf16.h>

typedef __attribute__((ext_vector_type(16))) int   v16i;
typedef __attribute__((ext_vector_type(8)))  float v8f;

#define FP8_MAX   448.0f
#define QUANT_EPS 1e-10f

__device__ __forceinline__ unsigned char f32_to_e4m3(float f) {
    unsigned int u    = __float_as_uint(f);
    unsigned int sign = (u >> 24) & 0x80u;
    unsigned int absu = u & 0x7FFFFFFFu;
    float        absf = __uint_as_float(absu);
    if (absu >= 0x7F800000u) return (unsigned char)(sign | 0x7Fu);
    if (absf < 0.015625f) {
        float r = rintf(absf * 512.0f);
        return (unsigned char)(sign | (unsigned int)r);
    }
    unsigned int rounded = absu + 0x7FFFFu + ((absu >> 20) & 1u);
    int          e       = (int)(rounded >> 23) - 127 + 7;
    unsigned int mant    = (rounded >> 20) & 7u;
    if (e > 15 || (e == 15 && mant == 7u)) return (unsigned char)(sign | 0x7Eu);
    return (unsigned char)(sign | ((unsigned int)e << 3) | mant);
}

__device__ __forceinline__ unsigned short f32_to_bf16_bits(float f) {
    unsigned u = __float_as_uint(f);
    unsigned r = u + 0x7FFFu + ((u >> 16) & 1u);
    return (unsigned short)(r >> 16);
}
__device__ __forceinline__ float bf16_rt(float f) {
    return __uint_as_float(((unsigned)f32_to_bf16_bits(f)) << 16);
}

__device__ __forceinline__ void async_b128(unsigned lds_off, unsigned long long gaddr) {
    asm volatile("global_load_async_to_lds_b128 %0, %1, off"
                 :: "v"(lds_off), "v"(gaddr) : "memory");
}

__global__ __launch_bounds__(256)
void quant_x_kernel(const __hip_bfloat16* __restrict__ x,
                    unsigned char* __restrict__ xq,
                    float* __restrict__ xs, int K) {
    const int m = blockIdx.x;
    const uint4* row = (const uint4*)(x + (size_t)m * K);
    const int nv = K >> 3;

    __shared__ float red[256];
    float amax = 0.0f;
    for (int i = threadIdx.x; i < nv; i += 256) {
        uint4 p = row[i];
        unsigned w[4] = {p.x, p.y, p.z, p.w};
        #pragma unroll
        for (int d = 0; d < 4; ++d) {
            float lo = __uint_as_float(w[d] << 16);
            float hi = __uint_as_float(w[d] & 0xFFFF0000u);
            amax = fmaxf(amax, fmaxf(fabsf(lo), fabsf(hi)));
        }
    }
    red[threadIdx.x] = amax;
    __syncthreads();
    for (int s = 128; s > 0; s >>= 1) {
        if (threadIdx.x < s)
            red[threadIdx.x] = fmaxf(red[threadIdx.x], red[threadIdx.x + s]);
        __syncthreads();
    }
    const float scale = fmaxf(red[0], QUANT_EPS) / FP8_MAX;
    const float inv   = 1.0f / scale;
    if (threadIdx.x == 0) xs[m] = scale;

    uint2* orow = (uint2*)(xq + (size_t)m * K);
    for (int i = threadIdx.x; i < nv; i += 256) {
        uint4 p = row[i];
        unsigned w[4] = {p.x, p.y, p.z, p.w};
        unsigned b[8];
        #pragma unroll
        for (int d = 0; d < 4; ++d) {
            b[2*d]   = f32_to_e4m3(__uint_as_float(w[d] << 16) * inv);
            b[2*d+1] = f32_to_e4m3(__uint_as_float(w[d] & 0xFFFF0000u) * inv);
        }
        uint2 o;
        o.x = b[0] | (b[1] << 8) | (b[2] << 16) | (b[3] << 24);
        o.y = b[4] | (b[5] << 8) | (b[6] << 16) | (b[7] << 24);
        orow[i] = o;
    }
}

__global__ __launch_bounds__(256)
void quant_w_kernel(const float* __restrict__ w,
                    unsigned char* __restrict__ wq, int K) {
    const int n = blockIdx.x;
    const float4* row  = (const float4*)(w + (size_t)n * K);
    unsigned*     orow = (unsigned*)(wq + (size_t)n * K);
    const int nv = K >> 2;
    for (int i = threadIdx.x; i < nv; i += 256) {
        float4 v = row[i];
        unsigned o = (unsigned)f32_to_e4m3(v.x)
                   | ((unsigned)f32_to_e4m3(v.y) << 8)
                   | ((unsigned)f32_to_e4m3(v.z) << 16)
                   | ((unsigned)f32_to_e4m3(v.w) << 24);
        orow[i] = o;
    }
}

__global__ __launch_bounds__(256)
void gemm_fp8_kernel(const unsigned char* __restrict__ Aq,
                     const unsigned char* __restrict__ Bq,
                     const float* __restrict__ xs,
                     const float* __restrict__ wsc,
                     const float* __restrict__ bias,
                     __hip_bfloat16* __restrict__ out,
                     int M, int N, int K) {
    constexpr int BM = 128, BN = 256, BK = 128;
    constexpr int ATILE = BM * BK;
    constexpr int BTILE = BN * BK;
    constexpr int BUFB  = ATILE + BTILE;

    __shared__ union {
        unsigned char  q[2][BUFB];
        unsigned short h[BM * BN];
    } sm;

    const int tid  = threadIdx.x;
    const int lane = tid & 31;
    const int wid  = tid >> 5;
    const int wm   = wid >> 2;
    const int wn   = wid & 3;
    const int l16  = lane & 15;
    const int lh   = lane >> 4;

    const int m0 = blockIdx.y * BM;
    const int n0 = blockIdx.x * BN;

    const int rowq = tid >> 3;
    const int coq  = (tid & 7) * 16;

    const unsigned smBase = (unsigned)(unsigned long long)(uintptr_t)&sm;
    const unsigned ldsA0  = smBase + (unsigned)(rowq * BK + coq);
    const unsigned ldsB0  = smBase + (unsigned)(ATILE + rowq * BK + coq);
    const unsigned long long aG =
        (unsigned long long)(uintptr_t)Aq + (unsigned long long)(m0 + rowq) * K + coq;
    const unsigned long long bG =
        (unsigned long long)(uintptr_t)Bq + (unsigned long long)(n0 + rowq) * K + coq;

    auto issue = [&](int buf, int k0) {
        const unsigned bo = (unsigned)buf * BUFB;
        #pragma unroll
        for (int i = 0; i < 4; ++i)
            async_b128(ldsA0 + bo + i * 32 * BK,
                       aG + (unsigned long long)k0 + (unsigned long long)i * 32u * (unsigned)K);
        #pragma unroll
        for (int i = 0; i < 8; ++i)
            async_b128(ldsB0 + bo + i * 32 * BK,
                       bG + (unsigned long long)k0 + (unsigned long long)i * 32u * (unsigned)K);
    };

    v8f acc[4][4] = {};

    auto compute = [&](int buf) {
        const unsigned char* lA = &sm.q[buf][0];
        const unsigned char* lB = &sm.q[buf][ATILE];

        union BF { v16i v; uint4 x[4]; unsigned long long q[8]; } bf[4];
        #pragma unroll
        for (int s = 0; s < 4; ++s) {
            const int col = wn * 64 + s * 16 + l16;
            #pragma unroll
            for (int g = 0; g < 8; ++g)
                bf[s].q[g] = *(const unsigned long long*)&lB[col * BK + g * 16 + lh * 8];
        }
        #pragma unroll
        for (int t = 0; t < 4; ++t) {
            union AF { v16i v; unsigned long long q[8]; } af;
            const int row = wm * 64 + t * 16 + l16;
            #pragma unroll
            for (int j = 0; j < 8; ++j)
                af.q[j] = *(const unsigned long long*)&lA[row * BK + j * 16 + lh * 8];
            #pragma unroll
            for (int s = 0; s < 4; ++s)
                acc[t][s] = __builtin_amdgcn_wmma_f32_16x16x128_fp8_fp8(
                    af.v, bf[s].v, (short)0, acc[t][s], false, false);
        }
    };

    const int nk = K / BK;
    issue(0, 0);

    for (int kk = 0; kk < nk - 1; ++kk) {
        issue((kk + 1) & 1, (kk + 1) * BK);
        asm volatile("s_wait_asynccnt 0xc" ::: "memory");
        __syncthreads();
        compute(kk & 1);
        __syncthreads();
    }
    asm volatile("s_wait_asynccnt 0x0" ::: "memory");
    __syncthreads();
    compute((nk - 1) & 1);
    __syncthreads();

    #pragma unroll
    for (int t = 0; t < 4; ++t) {
        const int rb = wm * 64 + t * 16;
        #pragma unroll
        for (int s = 0; s < 4; ++s) {
            const int   c   = wn * 64 + s * 16 + l16;
            const float wsn = wsc[n0 + c];
            const float bnb = bf16_rt(bias[n0 + c]);
            #pragma unroll
            for (int j = 0; j < 8; ++j) {
                const int r = rb + lh * 8 + j;
                float f = acc[t][s][j] * xs[m0 + r] * wsn;
                sm.h[r * BN + c] = f32_to_bf16_bits(bf16_rt(f) + bnb);
            }
        }
    }
    __syncthreads();

    #pragma unroll
    for (int i = 0; i < 16; ++i) {
        const int chunk = tid + i * 256;
        const int row   = chunk >> 5;
        const int co    = (chunk & 31) * 8;
        uint4 pk = *(const uint4*)&sm.h[row * BN + co];
        *(uint4*)&out[(size_t)(m0 + row) * N + n0 + co] = pk;
    }
}

extern "C" void kernel_launch(void* const* d_in, const int* in_sizes, int n_in,
                              void* d_out, int out_size, void* d_ws, size_t ws_size,
                              hipStream_t stream) {
    const __hip_bfloat16* x    = (const __hip_bfloat16*)d_in[0];
    const float*          w    = (const float*)d_in[1];
    const float*          wsc  = (const float*)d_in[2];
    const float*          bias = (const float*)d_in[3];
    __hip_bfloat16*       out  = (__hip_bfloat16*)d_out;

    const int N = in_sizes[2];
    const int K = in_sizes[1] / N;
    const int M = in_sizes[0] / K;

    unsigned char* xq  = (unsigned char*)d_ws;
    unsigned char* wq  = xq + (size_t)M * K;
    float*         xsc = (float*)(wq + (size_t)N * K);

    quant_x_kernel<<<M, 256, 0, stream>>>(x, xq, xsc, K);
    quant_w_kernel<<<N, 256, 0, stream>>>(w, wq, K);

    dim3 grid(N / 256, M / 128);
    gemm_fp8_kernel<<<grid, 256, 0, stream>>>(xq, wq, xsc, wsc, bias, out, M, N, K);
}
